// MultiLabelContrastiveLoss_16449724745562
// MI455X (gfx1250) — compile-verified, matrix operand order checked by reading, not run on hardware
//
#include <hip/hip_runtime.h>
#include <hip/hip_bf16.h>

#define NN 8192
#define DD 128
#define CC 28
#define INV_TEMP (1.0f / 0.07f)

// LDS staging geometry: 64 columns per stage, double buffered, hi+lo.
#define STAGE_COLS 64
#define LDS_STRIDE 136                       // 128 + 8 ushort pad -> 272B rows, 16B aligned
#define BUF_U      (STAGE_COLS * LDS_STRIDE) // ushorts per (buffer, array)
// Tropical row table T[b][r]: 28 x 128 floats, stride padded to 132 (528B rows).
#define T_STRIDE   132
#define T_OFF_U    (4u * BUF_U)              // ushort offset of T within smem (69,632 B, 16B aligned)
#define SMEM_BYTES (T_OFF_U * 2u + (unsigned)(CC * T_STRIDE * 4)) // 69,632 + 14,784 = 84,416 B

typedef __attribute__((ext_vector_type(16))) __bf16 v16bf;
typedef __attribute__((ext_vector_type(8)))  float  v8f;

// ---------------------------------------------------------------------------
// Prep: f32 -> bf16 hi/lo split (round-to-nearest-even, manual bit twiddling)
// ---------------------------------------------------------------------------
__global__ void prep_embed_kernel(const float* __restrict__ emb,
                                  unsigned short* __restrict__ hi,
                                  unsigned short* __restrict__ lo) {
    int idx = blockIdx.x * blockDim.x + threadIdx.x;
    if (idx >= NN * DD) return;
    float x = emb[idx];
    unsigned u = __float_as_uint(x);
    unsigned h = (u + 0x7FFFu + ((u >> 16) & 1u)) >> 16;
    hi[idx] = (unsigned short)h;
    float hf = __uint_as_float(h << 16);
    float r = x - hf;
    unsigned ur = __float_as_uint(r);
    unsigned l = (ur + 0x7FFFu + ((ur >> 16) & 1u)) >> 16;
    lo[idx] = (unsigned short)l;
}

// ---------------------------------------------------------------------------
// Prep: pack 28-bit label masks
// ---------------------------------------------------------------------------
__global__ void prep_labels_kernel(const int* __restrict__ labels,
                                   unsigned* __restrict__ masks) {
    int j = blockIdx.x * blockDim.x + threadIdx.x;
    if (j >= NN) return;
    unsigned mask = 0u;
    #pragma unroll
    for (int c = 0; c < CC; ++c)
        if (labels[j * CC + c] != 0) mask |= (1u << c);
    masks[j] = mask;
}

__global__ void init_accum_kernel(float* __restrict__ accum) {
    if (threadIdx.x < 2) accum[threadIdx.x] = 0.0f;
}

__global__ void final_kernel(const float* __restrict__ accum, float* __restrict__ out) {
    out[0] = accum[0] / fmaxf(accum[1], 1.0f);
}

// ---------------------------------------------------------------------------
// 16x32 bf16 A-fragment load from global (two B128 loads per lane, ISA layout)
// ---------------------------------------------------------------------------
__device__ __forceinline__ v16bf load_frag_global(const unsigned short* __restrict__ base,
                                                  int row0, int k0, int lane) {
    int r  = row0 + (lane & 15);
    int kk = k0 + ((lane >> 4) << 3);
    const unsigned short* p = base + r * DD + kk;
    union { uint4 u[2]; v16bf v; } U;
    U.u[0] = *(const uint4*)(p);
    U.u[1] = *(const uint4*)(p + 16);
    return U.v;
}

// Same fragment pattern but from the padded LDS stage buffer.
__device__ __forceinline__ v16bf load_frag_lds(const unsigned short* sh,
                                               int colbase, int k0, int lane) {
    int c  = colbase + (lane & 15);
    int kk = k0 + ((lane >> 4) << 3);
    const unsigned short* p = sh + c * LDS_STRIDE + kk;
    union { uint4 u[2]; v16bf v; } U;
    U.u[0] = *(const uint4*)(p);
    U.u[1] = *(const uint4*)(p + 16);
    return U.v;
}

__device__ __forceinline__ v8f wmma_bf16(v16bf a, v16bf b, v8f c) {
    return __builtin_amdgcn_wmma_f32_16x16x32_bf16(
        /*neg_a=*/false, a, /*neg_b=*/false, b,
        /*c_mod=*/(short)0, c, /*reuse_a=*/false, /*reuse_b=*/false);
}

// ---------------------------------------------------------------------------
// Async copy of one stage (64 cols x 128 k of one bf16 array) into LDS.
// 1024 x 16B chunks, 4 per thread; lane-consecutive chunks -> coalesced.
// Uses the CDNA5 async-DMA path (ASYNCcnt), no VGPR data movement.
// ---------------------------------------------------------------------------
__device__ __forceinline__ void stage_copy_async(const unsigned short* __restrict__ g,
                                                 unsigned lds_base, int jb, int tid) {
    #pragma unroll
    for (int q = 0; q < 4; ++q) {
        int chunk = q * 256 + tid;          // 0..1023
        int col   = chunk >> 4;             // 16 chunks per column
        int offu  = (chunk & 15) << 3;      // ushort offset within column
        const unsigned short* gp = g + (jb + col) * DD + offu;
        unsigned laddr = lds_base + (unsigned)(col * LDS_STRIDE + offu) * 2u;
        asm volatile("global_load_async_to_lds_b128 %0, %1, off"
                     :: "v"(laddr), "v"(gp) : "memory");
    }
}

// ---------------------------------------------------------------------------
// Main: block = 256 thr = 8 waves = 128 rows; waves share double-buffered
// LDS B-stages (async copy of stage s+1 overlaps compute on stage s).
// ---------------------------------------------------------------------------
__global__ __launch_bounds__(256)
void contrastive_main_kernel(const unsigned short* __restrict__ Ehi,
                             const unsigned short* __restrict__ Elo,
                             const unsigned* __restrict__ masks,
                             const float* __restrict__ va,
                             float* __restrict__ accum) {
    extern __shared__ unsigned short smem[];   // [hi b0][hi b1][lo b0][lo b1][T]
    const int tid  = threadIdx.x;
    const int lane = tid & 31;
    const int wave = tid >> 5;
    const int R0   = blockIdx.x * 128;         // block row base
    const int r0   = R0 + wave * 16;           // wave row base
    const int g    = lane >> 4;

    const unsigned lds_base = (unsigned)(size_t)(smem);
    float* Tlds = (float*)(smem + T_OFF_U);    // T[b][r], r block-local, stride 132

    // --- one-time per block: tropical row table T[b][r] = max_{a in mask_r} va[a][b]
    for (int idx = tid; idx < 128 * CC; idx += 256) {
        int r = idx & 127;
        int b = idx >> 7;
        unsigned mk = masks[R0 + r];
        float best = -1.0e30f;
        while (mk) {
            int a = __builtin_ctz(mk);
            mk &= mk - 1u;
            best = fmaxf(best, va[a * CC + b]);
        }
        Tlds[b * T_STRIDE + r] = best;
    }
    // (visible to all waves after the stage-0 barrier below)

    // A fragments (this wave's 16 rows x K=128, hi+lo) live in registers.
    v16bf ahi[4], alo[4];
    #pragma unroll
    for (int kc = 0; kc < 4; ++kc) {
        ahi[kc] = load_frag_global(Ehi, r0, kc * 32, lane);
        alo[kc] = load_frag_global(Elo, r0, kc * 32, lane);
    }

    unsigned rowmask[8];
    #pragma unroll
    for (int v = 0; v < 8; ++v) rowmask[v] = masks[r0 + g * 8 + v];

    float m[8], sp[8], sn[8];
    #pragma unroll
    for (int v = 0; v < 8; ++v) { m[v] = -1.0e4f; sp[v] = 0.0f; sn[v] = 0.0f; }

    const int NSTAGE = NN / STAGE_COLS;   // 128
    // Preload stage 0 into buffer 0.
    stage_copy_async(Ehi, lds_base + 0u * BUF_U * 2u, 0, tid);
    stage_copy_async(Elo, lds_base + 2u * BUF_U * 2u, 0, tid);

    for (int s = 0; s < NSTAGE; ++s) {
        const unsigned cur = (unsigned)(s & 1);
        asm volatile("s_wait_asynccnt 0x0" ::: "memory");
        __syncthreads();
        if (s + 1 < NSTAGE) {   // overlap: copy stage s+1 while computing stage s
            const unsigned nxt = 1u - cur;
            stage_copy_async(Ehi, lds_base + nxt        * BUF_U * 2u, (s + 1) * STAGE_COLS, tid);
            stage_copy_async(Elo, lds_base + (2u + nxt) * BUF_U * 2u, (s + 1) * STAGE_COLS, tid);
        }
        const unsigned short* shHi = smem + cur * BUF_U;
        const unsigned short* shLo = smem + (2u + cur) * BUF_U;

        #pragma unroll
        for (int jt = 0; jt < STAGE_COLS / 16; ++jt) {
            v8f c = {};
            #pragma unroll
            for (int kc = 0; kc < 4; ++kc) {
                v16bf bhi = load_frag_lds(shHi, jt * 16, kc * 32, lane);
                v16bf blo = load_frag_lds(shLo, jt * 16, kc * 32, lane);
                c = wmma_bf16(ahi[kc], bhi, c);   // hi*hi
                c = wmma_bf16(ahi[kc], blo, c);   // hi*lo
                c = wmma_bf16(alo[kc], bhi, c);   // lo*hi (lo*lo dropped: ~2^-16)
            }

            const int j = s * STAGE_COLS + jt * 16 + (lane & 15);
            const unsigned mj = masks[j];

            // W0 for all 8 rows at once: one bit-loop over mask_j with
            // two ds_load_b128 of T[b][rowbase..rowbase+7] per set bit.
            float w0v[8];
            #pragma unroll
            for (int v = 0; v < 8; ++v) w0v[v] = -1.0e30f;
            {
                const float* tp = Tlds + (wave * 16 + g * 8);
                unsigned mm = mj;
                while (mm) {
                    int b = __builtin_ctz(mm);
                    mm &= mm - 1u;
                    const float4* p = (const float4*)(tp + b * T_STRIDE);
                    float4 t0 = p[0];
                    float4 t1 = p[1];
                    w0v[0] = fmaxf(w0v[0], t0.x);
                    w0v[1] = fmaxf(w0v[1], t0.y);
                    w0v[2] = fmaxf(w0v[2], t0.z);
                    w0v[3] = fmaxf(w0v[3], t0.w);
                    w0v[4] = fmaxf(w0v[4], t1.x);
                    w0v[5] = fmaxf(w0v[5], t1.y);
                    w0v[6] = fmaxf(w0v[6], t1.z);
                    w0v[7] = fmaxf(w0v[7], t1.w);
                }
            }

            #pragma unroll
            for (int v = 0; v < 8; ++v) {
                const int i = r0 + g * 8 + v;
                float x = c[v] * INV_TEMP;
                if (i == j) x = -__builtin_inff();   // exclude diagonal
                const bool shared = (rowmask[v] & mj) != 0u;
                const float w = shared ? 1.0f : w0v[v];
                const float mn    = fmaxf(m[v], x);
                const float scale = __expf(m[v] - mn);
                const float e     = __expf(x - mn);
                sn[v] = sn[v] * scale + e * w;
                sp[v] = sp[v] * scale + (shared ? e : 0.0f);
                m[v]  = mn;
            }
        }
    }

    // Merge the 16 column-lanes of each half-wave (softmax-merge butterfly).
    #pragma unroll
    for (int v = 0; v < 8; ++v) {
        float mv = m[v], spv = sp[v], snv = sn[v];
        #pragma unroll
        for (int off = 1; off < 16; off <<= 1) {
            float m2  = __shfl_xor(mv,  off, 16);
            float sp2 = __shfl_xor(spv, off, 16);
            float sn2 = __shfl_xor(snv, off, 16);
            float M   = fmaxf(mv, m2);
            float a0  = __expf(mv - M);
            float a1  = __expf(m2 - M);
            spv = spv * a0 + sp2 * a1;
            snv = snv * a0 + sn2 * a1;
            mv  = M;
        }
        if ((lane & 15) == 0) {
            if (spv > 0.0f) {  // has_pos <=> pos_exp > 0
                float loss = -__logf(spv / (snv + 1e-8f));
                atomicAdd(&accum[0], loss);
                atomicAdd(&accum[1], 1.0f);
            }
        }
    }
}

// ---------------------------------------------------------------------------
extern "C" void kernel_launch(void* const* d_in, const int* in_sizes, int n_in,
                              void* d_out, int out_size, void* d_ws, size_t ws_size,
                              hipStream_t stream) {
    const float* emb    = (const float*)d_in[0];   // (N, D) f32
    const int*   labels = (const int*)d_in[1];     // (N, C) i32
    const float* va     = (const float*)d_in[2];   // (C, C) f32
    float*       out    = (float*)d_out;

    char* ws = (char*)d_ws;
    unsigned short* Ehi   = (unsigned short*)(ws);                        // 2 MB
    unsigned short* Elo   = (unsigned short*)(ws + 2u * 1024 * 1024);     // 2 MB
    unsigned*       masks = (unsigned*)(ws + 4u * 1024 * 1024);           // 32 KB
    float*          accum = (float*)(ws + 4u * 1024 * 1024 + 64u * 1024); // 2 floats

    init_accum_kernel<<<1, 32, 0, stream>>>(accum);
    prep_embed_kernel<<<(NN * DD + 255) / 256, 256, 0, stream>>>(emb, Ehi, Elo);
    prep_labels_kernel<<<(NN + 255) / 256, 256, 0, stream>>>(labels, masks);
    contrastive_main_kernel<<<NN / 128, 256, SMEM_BYTES, stream>>>(Ehi, Elo, masks, va, accum);
    final_kernel<<<1, 1, 0, stream>>>(accum, out);
}
